// LightGCN_84430467105382
// MI455X (gfx1250) — matrix operand order repaired and checked by reading, not compiled, not run on hardware
//
#include <hip/hip_runtime.h>
#include <hip/hip_bf16.h>

typedef __attribute__((ext_vector_type(16))) _Float16 v16h;
typedef __attribute__((ext_vector_type(8)))  float    v8f;
typedef __attribute__((ext_vector_type(8)))  unsigned v8u;

#define ND_C   5000
#define NND    12000            // total nodes (5000 diseases + 7000 drugs)
#define DIM    64               // embedding dim
#define KWORDS 376              // bitmap dwords per row -> padded K = 12032
#define ZPAD   12032            // padded K (columns of z_t)
#define KSPLIT 8
#define KB_PER (KWORDS / KSPLIT)  // 47 K-blocks of 32 per chunk

// ---------------- zero fill (bitmap) ----------------
__global__ void zero_kernel(uint4* __restrict__ p, int n4) {
    int i = blockIdx.x * blockDim.x + threadIdx.x;
    if (i < n4) p[i] = make_uint4(0u, 0u, 0u, 0u);
}

// ---------------- scatter edges into bitmap ----------------
__global__ void build_kernel(const int* __restrict__ ei, unsigned* __restrict__ bitmap, int E) {
    int e = blockIdx.x * blockDim.x + threadIdx.x;
    if (e >= E) return;
    int s = ei[e];
    int d = ei[E + e];
    atomicOr(bitmap + (size_t)s * KWORDS + (d >> 5), 1u << (d & 31));
    atomicOr(bitmap + (size_t)d * KWORDS + (s >> 5), 1u << (s & 31));
}

// ---------------- degrees -> dinv = 1/sqrt(deg + 1e-12) ----------------
__global__ void degree_kernel(const unsigned* __restrict__ bitmap, float* __restrict__ dinv) {
    int i = blockIdx.x * blockDim.x + threadIdx.x;
    if (i >= NND) return;
    const uint4* row = (const uint4*)(bitmap + (size_t)i * KWORDS);
    int deg = 0;
    for (int k = 0; k < KWORDS / 4; ++k) {
        uint4 w = row[k];
        deg += __popc(w.x) + __popc(w.y) + __popc(w.z) + __popc(w.w);
    }
    dinv[i] = 1.0f / sqrtf((float)deg + 1e-12f);
}

// ---------------- prep: xsum = all_emb ; z_t[d][i] = f16(dinv[i]*all_emb[i,d]) ----------------
__global__ void prep_kernel(const float* __restrict__ dis, const float* __restrict__ drug,
                            const float* __restrict__ dinv,
                            float* __restrict__ xsum, _Float16* __restrict__ zt) {
    int idx = blockIdx.x * blockDim.x + threadIdx.x;
    if (idx >= DIM * ZPAD) return;
    int d = idx / ZPAD;
    int i = idx - d * ZPAD;
    if (i < NND) {
        float e = (i < ND_C) ? dis[(size_t)i * DIM + d] : drug[(size_t)(i - ND_C) * DIM + d];
        xsum[(size_t)i * DIM + d] = e;
        zt[idx] = (_Float16)(dinv[i] * e);
    } else {
        zt[idx] = (_Float16)0.f;
    }
}

// ---------------- WMMA SpMM over bitmap: spart[c] = A_bits(chunk c) @ z ----------------
// One wave per (16-row tile, K-chunk). C tiles: 16 rows x 64 cols (4x v8f).
// Software-pipelined: iteration kb issues loads for kb+1 before kb's WMMAs.
// A-fragment expansion via 4KB LDS LUT: byte -> 4 dwords of packed f16 {0,1} pairs.
__global__ __launch_bounds__(32) void spmm_kernel(const unsigned* __restrict__ bitmap,
                                                  const _Float16* __restrict__ zt,
                                                  float* __restrict__ spart) {
    __shared__ unsigned lut[256][4];
    const int lane = threadIdx.x;          // 0..31

    // Build LUT: entry e, dword v holds f16 pair from bits (2v, 2v+1) of e.
#pragma unroll
    for (int e8 = 0; e8 < 8; ++e8) {
        int e = lane + 32 * e8;
#pragma unroll
        for (int v = 0; v < 4; ++v) {
            unsigned lo = (unsigned)(e >> (2 * v)) & 1u;
            unsigned hi = (unsigned)(e >> (2 * v + 1)) & 1u;
            lut[e][v] = (lo ? 0x00003C00u : 0u) | (hi ? 0x3C000000u : 0u);
        }
    }
    __syncthreads();   // single-wave workgroup: lowers to waitcnt/nop

    const int m    = lane & 15;            // row within tile / B column
    const int hsel = lane >> 4;
    const int r    = blockIdx.x;           // row tile (750)
    const int c    = blockIdx.y;           // K chunk (8)

    const unsigned* wrow  = bitmap + (size_t)(16 * r + m) * KWORDS;
    const _Float16* zbase = zt + (size_t)m * ZPAD + 16 * hsel;

    v8f c0 = {}, c1 = {}, c2 = {}, c3 = {};

    const int kb0 = c * KB_PER, kbend = kb0 + KB_PER;

    // prologue: loads for first K-block
    unsigned w = wrow[kb0];
    const _Float16* zb = zbase + (size_t)kb0 * 32;
    v16h b0 = *(const v16h*)(zb);
    v16h b1 = *(const v16h*)(zb + 16 * ZPAD);
    v16h b2 = *(const v16h*)(zb + 32 * ZPAD);
    v16h b3 = *(const v16h*)(zb + 48 * ZPAD);

    for (int kb = kb0; kb < kbend - 1; ++kb) {
        // ---- prefetch next iteration (fresh registers; hides L2 latency) ----
        unsigned wn = wrow[kb + 1];
        const _Float16* zn = zbase + (size_t)(kb + 1) * 32;
        v16h n0 = *(const v16h*)(zn);
        v16h n1 = *(const v16h*)(zn + 16 * ZPAD);
        v16h n2 = *(const v16h*)(zn + 32 * ZPAD);
        v16h n3 = *(const v16h*)(zn + 48 * ZPAD);

        // ---- expand current 16x32 bit tile via LDS LUT ----
        unsigned ws = hsel ? (w >> 16) : w;
        uint4 alo = *(const uint4*)lut[ws & 0xFFu];          // ds_load_b128
        uint4 ahi = *(const uint4*)lut[(ws >> 8) & 0xFFu];   // ds_load_b128
        v8u au;
        au[0] = alo.x; au[1] = alo.y; au[2] = alo.z; au[3] = alo.w;
        au[4] = ahi.x; au[5] = ahi.y; au[6] = ahi.z; au[7] = ahi.w;
        v16h a = __builtin_bit_cast(v16h, au);

        c0 = __builtin_amdgcn_wmma_f32_16x16x32_f16(false, a, false, b0, (short)0, c0, false, false);
        c1 = __builtin_amdgcn_wmma_f32_16x16x32_f16(false, a, false, b1, (short)0, c1, false, false);
        c2 = __builtin_amdgcn_wmma_f32_16x16x32_f16(false, a, false, b2, (short)0, c2, false, false);
        c3 = __builtin_amdgcn_wmma_f32_16x16x32_f16(false, a, false, b3, (short)0, c3, false, false);

        // rotate pipeline
        w = wn; b0 = n0; b1 = n1; b2 = n2; b3 = n3;
    }

    // ---- epilogue: last K-block ----
    {
        unsigned ws = hsel ? (w >> 16) : w;
        uint4 alo = *(const uint4*)lut[ws & 0xFFu];
        uint4 ahi = *(const uint4*)lut[(ws >> 8) & 0xFFu];
        v8u au;
        au[0] = alo.x; au[1] = alo.y; au[2] = alo.z; au[3] = alo.w;
        au[4] = ahi.x; au[5] = ahi.y; au[6] = ahi.z; au[7] = ahi.w;
        v16h a = __builtin_bit_cast(v16h, au);

        c0 = __builtin_amdgcn_wmma_f32_16x16x32_f16(false, a, false, b0, (short)0, c0, false, false);
        c1 = __builtin_amdgcn_wmma_f32_16x16x32_f16(false, a, false, b1, (short)0, c1, false, false);
        c2 = __builtin_amdgcn_wmma_f32_16x16x32_f16(false, a, false, b2, (short)0, c2, false, false);
        c3 = __builtin_amdgcn_wmma_f32_16x16x32_f16(false, a, false, b3, (short)0, c3, false, false);
    }

    // ---- write partial: C layout -> row = 16r + 8*hsel + p, col = 16t + m ----
    float* srow = spart + (size_t)c * ((size_t)NND * DIM)
                        + (size_t)(16 * r + 8 * hsel) * DIM + m;
#pragma unroll
    for (int p = 0; p < 8; ++p) {
        srow[p * DIM +  0] = c0[p];
        srow[p * DIM + 16] = c1[p];
        srow[p * DIM + 32] = c2[p];
        srow[p * DIM + 48] = c3[p];
    }
}

// ---------------- post: x = dinv * sum(partials); xsum += x; z_t = f16(dinv*x) ----------------
__global__ void post_kernel(const float* __restrict__ sp, const float* __restrict__ dinv,
                            float* __restrict__ xsum, _Float16* __restrict__ zt) {
    int idx = blockIdx.x * blockDim.x + threadIdx.x;
    if (idx >= DIM * ZPAD) return;
    int d = idx / ZPAD;
    int i = idx - d * ZPAD;
    if (i < NND) {
        float t = 0.f;
#pragma unroll
        for (int c = 0; c < KSPLIT; ++c)
            t += sp[(size_t)c * ((size_t)NND * DIM) + (size_t)i * DIM + d];
        float di = dinv[i];
        float v  = di * t;
        xsum[(size_t)i * DIM + d] += v;
        zt[idx] = (_Float16)(di * v);
    } else {
        zt[idx] = (_Float16)0.f;
    }
}

// ---------------- final: mean over (1 + L) = 4 stages ----------------
__global__ void final_kernel(const float* __restrict__ xsum, float* __restrict__ out) {
    int idx = blockIdx.x * blockDim.x + threadIdx.x;
    if (idx < NND * DIM) out[idx] = xsum[idx] * 0.25f;
}

extern "C" void kernel_launch(void* const* d_in, const int* in_sizes, int n_in,
                              void* d_out, int out_size, void* d_ws, size_t ws_size,
                              hipStream_t stream) {
    const float* dis  = (const float*)d_in[0];   // [5000,64]
    const float* drug = (const float*)d_in[1];   // [7000,64]
    const int*   ei   = (const int*)d_in[2];     // [2,E]
    const int    E    = in_sizes[2] / 2;
    float* out = (float*)d_out;

    // workspace carve-out (256B aligned): ~48 MB
    char* ws = (char*)d_ws;
    size_t off = 0;
    auto carve = [&](size_t bytes) -> void* {
        void* p = ws + off;
        off = (off + bytes + 255) & ~(size_t)255;
        return p;
    };
    unsigned* bitmap = (unsigned*)carve((size_t)NND * KWORDS * 4);          // 18.0 MB
    float*    dinv   = (float*)   carve((size_t)NND * 4);                   // 48 KB
    float*    xsum   = (float*)   carve((size_t)NND * DIM * 4);             //  3 MB
    float*    spart  = (float*)   carve((size_t)KSPLIT * NND * DIM * 4);    // 24.6 MB
    _Float16* zt     = (_Float16*)carve((size_t)DIM * ZPAD * 2);            // 1.5 MB

    const int nbits4 = (NND * KWORDS) / 4;
    zero_kernel<<<(nbits4 + 255) / 256, 256, 0, stream>>>((uint4*)bitmap, nbits4);
    build_kernel<<<(E + 255) / 256, 256, 0, stream>>>(ei, bitmap, E);
    degree_kernel<<<(NND + 255) / 256, 256, 0, stream>>>(bitmap, dinv);
    prep_kernel<<<(DIM * ZPAD + 255) / 256, 256, 0, stream>>>(dis, drug, dinv, xsum, zt);

    for (int l = 0; l < 3; ++l) {
        spmm_kernel<<<dim3(NND / 16, KSPLIT), 32, 0, stream>>>(bitmap, zt, spart);
        post_kernel<<<(DIM * ZPAD + 255) / 256, 256, 0, stream>>>(spart, dinv, xsum, zt);
    }
    final_kernel<<<(NND * DIM + 255) / 256, 256, 0, stream>>>(xsum, out);
}
